// MatchModel_52037823758578
// MI455X (gfx1250) — matrix operand order repaired and checked by reading, not compiled, not run on hardware
//
#include <hip/hip_runtime.h>
#include <hip/hip_bf16.h>
#include <math.h>

typedef _Float16 v16h __attribute__((ext_vector_type(16)));
typedef _Float16 v8h  __attribute__((ext_vector_type(8)));
typedef float    v8f  __attribute__((ext_vector_type(8)));

#define EMB   300
#define HID   256
#define TT    50
#define BATCH 64
#define PER   50
#define NCTX  (BATCH * PER)      // 3200
#define NTOT  (NCTX + BATCH)     // 3264 = 204 * 16 exactly
#define KX    320                // EMB padded to 10 K-chunks of 32
#define KH    256                // 8 K-chunks of 32
#define G3    (3 * HID)          // 768 gate columns
#define HSTR  264                // LDS h stride (halves), 528B = 33*16B aligned

// ---------------------------------------------------------------------------
// WMMA fragment loaders (16x16x32 f16; this code's contiguous k order, same in A and B)
// A (16xK tile, row-major, stride in halves):
//   lane<16 : row=lane,    K = k0..k0+15
//   lane>=16: row=lane-16, K = k0+16..k0+31
__device__ inline v16h frag_a(const _Float16* base, int stride, int lane) {
  int row = lane & 15;
  int kb  = (lane >> 4) << 4;            // 0 or 16 halves
  const _Float16* p = base + (size_t)row * stride + kb;
  v8h lo = *(const v8h*)(p);             // ds/global b128
  v8h hi = *(const v8h*)(p + 8);         // ds/global b128
  return __builtin_shufflevector(lo, hi, 0, 1, 2, 3, 4, 5, 6, 7,
                                 8, 9, 10, 11, 12, 13, 14, 15);
}
// B (Kx16 tile) from weights stored row-major [N][K] (f16). Column n of B is
// row n of W, so each lane reads 32 CONTIGUOUS bytes -> one v16h load.
//   lane<16 : n=lane,    K = k0..k0+15 ; lane>=16: n=lane-16, K = k0+16..k0+31
__device__ inline v16h frag_b(const _Float16* W, int strideK, int n0, int k0, int lane) {
  int n  = lane & 15;
  int kb = (lane >> 4) << 4;             // 0 or 16 halves
  return *(const v16h*)(W + (size_t)(n0 + n) * strideK + k0 + kb);
}
__device__ inline v8f wmma_f16(v16h a, v16h b, v8f c) {
  return __builtin_amdgcn_wmma_f32_16x16x32_f16(false, a, false, b, (short)0, c,
                                                false, false);
}
__device__ inline float sigm(float x) { return 1.0f / (1.0f + expf(-x)); }

// ---------------------------------------------------------------------------
// Weight convert: f32 [rows][scols] -> f16 [rows][dcols], zero padded
__global__ void cvt_kernel(const float* __restrict__ src, _Float16* __restrict__ dst,
                           int rows, int scols, int dcols) {
  int idx = blockIdx.x * 256 + threadIdx.x;
  if (idx >= rows * dcols) return;
  int r = idx / dcols, c = idx % dcols;
  dst[idx] = (c < scols) ? (_Float16)src[(size_t)r * scols + c] : (_Float16)0.0f;
}

// ---------------------------------------------------------------------------
// Embedding gather -> f16, padded to KX. Row r<NCTX = context, else message.
__global__ __launch_bounds__(256) void embed_kernel(
    const int* __restrict__ ctx, const int* __restrict__ msg,
    const float* __restrict__ emb, _Float16* __restrict__ Xf) {
  int r = blockIdx.x;
  const int* toks = (r < NCTX) ? (ctx + (size_t)r * TT)
                               : (msg + (size_t)(r - NCTX) * TT);
  for (int t = 0; t < TT; ++t) {
    int tok = toks[t];
    const float* e = emb + (size_t)tok * EMB;
    _Float16* d = Xf + ((size_t)r * TT + t) * KX;
    for (int c = threadIdx.x; c < KX; c += 256)
      d[c] = (c < EMB) ? (_Float16)e[c] : (_Float16)0.0f;
  }
}

// ---------------------------------------------------------------------------
// GRU recurrence. One block = 16 sequences x one direction, scans all T steps
// with h[16,256] resident in LDS. 8 waves; wave w owns hidden units
// [32w, 32w+32) -> gate column tiles {j0, j0+16} + {256,512} offsets, so all
// three gates of a unit share the same lane/VGPR slot of the WMMA D layout.
// K-chunk loops kept ROLLED to bound register pressure (weights stream from
// L2 each iteration; ~120 live VGPRs -> no spills, good occupancy).
__global__ __launch_bounds__(256) void gru_kernel(
    const _Float16* __restrict__ Xf,
    const _Float16* __restrict__ WihF, const _Float16* __restrict__ WhhF,
    const _Float16* __restrict__ WihB, const _Float16* __restrict__ WhhB,
    const float* __restrict__ bihF, const float* __restrict__ bhhF,
    const float* __restrict__ bihB, const float* __restrict__ bhhB,
    const int* __restrict__ ctx_len, const int* __restrict__ msg_len,
    _Float16* __restrict__ hcat) {
  __shared__ __align__(16) _Float16 xs[16 * KX];    // 10 KB  x_t tile
  __shared__ __align__(16) _Float16 hsh[16 * HSTR]; // 8.25 KB h state
  __shared__ int lens[16];

  const int tid = threadIdx.x, lane = tid & 31, wv = tid >> 5;
  const int dir = blockIdx.x & 1, s0 = (blockIdx.x >> 1) * 16;
  const _Float16* Wih = dir ? WihB : WihF;
  const _Float16* Whh = dir ? WhhB : WhhF;
  const float* bih = dir ? bihB : bihF;
  const float* bhh = dir ? bhhB : bhhF;

  if (tid < 16) {
    int s = s0 + tid;
    lens[tid] = (s < NCTX) ? ctx_len[s] : msg_len[s - NCTX];
  }
  for (int i = tid; i < 16 * HSTR; i += 256) hsh[i] = (_Float16)0.0f;

  const int j0 = wv * 32;        // wave's unit-tile base (+u*16)
  const int jc = lane & 15;      // column within tile
  float bR[2], bZ[2], bIN[2], bHN[2];
#pragma unroll
  for (int u = 0; u < 2; ++u) {
    int j = j0 + u * 16 + jc;
    bR[u]  = bih[j] + bhh[j];
    bZ[u]  = bih[HID + j] + bhh[HID + j];
    bIN[u] = bih[2 * HID + j];
    bHN[u] = bhh[2 * HID + j];
  }

  for (int t = 0; t < TT; ++t) {
    // Stage x_t (or reversed x for backward dir) into LDS: 640 x 16B chunks.
    for (int c = tid; c < 16 * 40; c += 256) {
      int row = c / 40, off = (c % 40) * 8;
      int len = lens[row];
      int te  = dir ? ((len - 1 - t) > 0 ? (len - 1 - t) : 0) : t;
      *(v8h*)(&xs[row * KX + off]) =
          *(const v8h*)(Xf + ((size_t)(s0 + row) * TT + te) * KX + off);
    }
    __syncthreads();  // xs ready; also fences previous step's h writes

    v8f aR[2] = {}, aZ[2] = {}, aIN[2] = {}, aHN[2] = {};
    // gates += x_t @ Wih^T   (r, z, i_n)
#pragma clang loop unroll(disable)
    for (int kc = 0; kc < KX / 32; ++kc) {
      v16h a = frag_a(xs + kc * 32, KX, lane);
#pragma unroll
      for (int u = 0; u < 2; ++u) {
        int n = j0 + u * 16;
        aR[u]  = wmma_f16(a, frag_b(Wih, KX, n,           kc * 32, lane), aR[u]);
        aZ[u]  = wmma_f16(a, frag_b(Wih, KX, HID + n,     kc * 32, lane), aZ[u]);
        aIN[u] = wmma_f16(a, frag_b(Wih, KX, 2 * HID + n, kc * 32, lane), aIN[u]);
      }
    }
    // gates += h @ Whh^T     (r, z, h_n)
#pragma clang loop unroll(disable)
    for (int kc = 0; kc < KH / 32; ++kc) {
      v16h a = frag_a(hsh + kc * 32, HSTR, lane);
#pragma unroll
      for (int u = 0; u < 2; ++u) {
        int n = j0 + u * 16;
        aR[u]  = wmma_f16(a, frag_b(Whh, KH, n,           kc * 32, lane), aR[u]);
        aZ[u]  = wmma_f16(a, frag_b(Whh, KH, HID + n,     kc * 32, lane), aZ[u]);
        aHN[u] = wmma_f16(a, frag_b(Whh, KH, 2 * HID + n, kc * 32, lane), aHN[u]);
      }
    }
    __syncthreads();  // all h reads for this step complete

    // GRU cell update (D layout: lane -> col j, VGPR r -> row (lane>>4)*8 + r)
#pragma unroll
    for (int u = 0; u < 2; ++u) {
      int j = j0 + u * 16 + jc;
#pragma unroll
      for (int r = 0; r < 8; ++r) {
        int m = ((lane >> 4) << 3) + r;
        float hp = (float)hsh[m * HSTR + j];
        float rg = sigm(aR[u][r] + bR[u]);
        float zg = sigm(aZ[u][r] + bZ[u]);
        float cg = tanhf(aIN[u][r] + bIN[u] + rg * (aHN[u][r] + bHN[u]));
        float hv = (t < lens[m]) ? ((1.0f - zg) * cg + zg * hp) : hp;
        hsh[m * HSTR + j] = (_Float16)hv;
      }
    }
    __syncthreads();
  }

  // h -> hcat[:, dir*HID : dir*HID+HID]
  for (int c = tid; c < 16 * 16; c += 256) {
    int row = c >> 4, ch = c & 15;
    const v8h* s = (const v8h*)(&hsh[row * HSTR + ch * 16]);
    _Float16* d = hcat + (size_t)(s0 + row) * (2 * HID) + dir * HID + ch * 16;
    *(v8h*)(d) = s[0];
    *(v8h*)(d + 8) = s[1];
  }
}

// ---------------------------------------------------------------------------
// enc[3264,256] = hcat[3264,512](f16) @ Wout^T(f16) + b_out  (WMMA GEMM)
__global__ __launch_bounds__(256) void outproj_kernel(
    const _Float16* __restrict__ hcat, const _Float16* __restrict__ WoutH,
    const float* __restrict__ bout, float* __restrict__ enc) {
  const int tid = threadIdx.x, lane = tid & 31, wv = tid >> 5;
  const int m0 = blockIdx.x * 16;
  const int n0 = wv * 32;
  v8f acc[2] = {};
#pragma clang loop unroll(disable)
  for (int kc = 0; kc < (2 * HID) / 32; ++kc) {
    v16h a = frag_a(hcat + (size_t)m0 * (2 * HID) + kc * 32, 2 * HID, lane);
#pragma unroll
    for (int u = 0; u < 2; ++u)
      acc[u] = wmma_f16(a, frag_b(WoutH, 2 * HID, n0 + u * 16, kc * 32, lane), acc[u]);
  }
#pragma unroll
  for (int u = 0; u < 2; ++u) {
    int n = n0 + u * 16 + (lane & 15);
    float bb = bout[n];
#pragma unroll
    for (int r = 0; r < 8; ++r) {
      int m = m0 + ((lane >> 4) << 3) + r;
      enc[(size_t)m * HID + n] = acc[u][r] + bb;
    }
  }
}

// ---------------------------------------------------------------------------
// Attention scores: w[i] = enc[i] . w_att + b_att   (one wave per row)
__global__ __launch_bounds__(256) void att_score_kernel(
    const float* __restrict__ enc, const float* __restrict__ watt,
    const float* __restrict__ batt, float* __restrict__ score) {
  int wv = threadIdx.x >> 5, lane = threadIdx.x & 31;
  int i = blockIdx.x * 8 + wv;
  if (i >= NCTX) return;
  float s = 0.0f;
#pragma unroll
  for (int e = 0; e < HID / 32; ++e) {
    int c = e * 32 + lane;
    s += enc[(size_t)i * HID + c] * watt[c];
  }
#pragma unroll
  for (int off = 16; off > 0; off >>= 1) s += __shfl_xor(s, off, 32);
  if (lane == 0) score[i] = s + batt[0];
}

// ---------------------------------------------------------------------------
// Segment softmax pooling + final logit. One block per batch element.
__global__ __launch_bounds__(256) void pool_kernel(
    const float* __restrict__ enc, const float* __restrict__ score,
    const int* __restrict__ clen, float* __restrict__ out) {
  __shared__ float red[256];
  const int b = blockIdx.x, tid = threadIdx.x;
  int start = 0;
  for (int i = 0; i < b; ++i) start += clen[i];
  const int cnt = clen[b];

  float mx = -1e30f;
  for (int i = tid; i < cnt; i += 256) mx = fmaxf(mx, score[start + i]);
  red[tid] = mx; __syncthreads();
  for (int s = 128; s > 0; s >>= 1) {
    if (tid < s) red[tid] = fmaxf(red[tid], red[tid + s]);
    __syncthreads();
  }
  mx = red[0]; __syncthreads();

  float sm = 0.0f;
  for (int i = tid; i < cnt; i += 256) sm += expf(score[start + i] - mx);
  red[tid] = sm; __syncthreads();
  for (int s = 128; s > 0; s >>= 1) {
    if (tid < s) red[tid] += red[tid + s];
    __syncthreads();
  }
  float denom = red[0]; __syncthreads();

  // thread tid owns hidden column tid (blockDim == HID)
  float pj = 0.0f;
  for (int i = 0; i < cnt; ++i)
    pj += expf(score[start + i] - mx) * enc[(size_t)(start + i) * HID + tid];
  pj /= denom;
  red[tid] = pj * enc[(size_t)(NCTX + b) * HID + tid];
  __syncthreads();
  for (int s = 128; s > 0; s >>= 1) {
    if (tid < s) red[tid] += red[tid + s];
    __syncthreads();
  }
  if (tid == 0) {
    float pos = sigm(red[0]);
    out[2 * b + 0] = 1.0f - pos;
    out[2 * b + 1] = pos;
  }
}

// ---------------------------------------------------------------------------
extern "C" void kernel_launch(void* const* d_in, const int* in_sizes, int n_in,
                              void* d_out, int out_size, void* d_ws, size_t ws_size,
                              hipStream_t stream) {
  const int*   context = (const int*)d_in[0];
  const int*   ctx_len = (const int*)d_in[1];
  const int*   seg_len = (const int*)d_in[2];
  const int*   message = (const int*)d_in[3];
  const int*   msg_len = (const int*)d_in[4];
  const float* emb     = (const float*)d_in[5];
  const float* w_ih_f  = (const float*)d_in[6];
  const float* w_hh_f  = (const float*)d_in[7];
  const float* b_ih_f  = (const float*)d_in[8];
  const float* b_hh_f  = (const float*)d_in[9];
  const float* w_ih_b  = (const float*)d_in[10];
  const float* w_hh_b  = (const float*)d_in[11];
  const float* b_ih_b  = (const float*)d_in[12];
  const float* b_hh_b  = (const float*)d_in[13];
  const float* w_out   = (const float*)d_in[14];
  const float* b_out   = (const float*)d_in[15];
  const float* w_att   = (const float*)d_in[16];
  const float* b_att   = (const float*)d_in[17];
  float* out = (float*)d_out;

  char* ws = (char*)d_ws;
  size_t off = 0;
  auto alloc = [&](size_t bytes) -> void* {
    void* p = ws + off;
    off += (bytes + 255) & ~(size_t)255;
    return p;
  };
  _Float16* Xf    = (_Float16*)alloc((size_t)NTOT * TT * KX * 2);  // ~104 MB
  _Float16* hcat  = (_Float16*)alloc((size_t)NTOT * 2 * HID * 2);
  float*    enc   = (float*)   alloc((size_t)NTOT * HID * 4);
  float*    sc    = (float*)   alloc((size_t)NCTX * 4);
  _Float16* WihF  = (_Float16*)alloc((size_t)G3 * KX * 2);
  _Float16* WihB  = (_Float16*)alloc((size_t)G3 * KX * 2);
  _Float16* WhhF  = (_Float16*)alloc((size_t)G3 * KH * 2);
  _Float16* WhhB  = (_Float16*)alloc((size_t)G3 * KH * 2);
  _Float16* WoutH = (_Float16*)alloc((size_t)HID * 2 * HID * 2);

  cvt_kernel<<<(G3 * KX + 255) / 256, 256, 0, stream>>>(w_ih_f, WihF, G3, EMB, KX);
  cvt_kernel<<<(G3 * KX + 255) / 256, 256, 0, stream>>>(w_ih_b, WihB, G3, EMB, KX);
  cvt_kernel<<<(G3 * KH + 255) / 256, 256, 0, stream>>>(w_hh_f, WhhF, G3, HID, KH);
  cvt_kernel<<<(G3 * KH + 255) / 256, 256, 0, stream>>>(w_hh_b, WhhB, G3, HID, KH);
  cvt_kernel<<<(HID * 2 * HID + 255) / 256, 256, 0, stream>>>(w_out, WoutH, HID, 2 * HID, 2 * HID);

  embed_kernel<<<NTOT, 256, 0, stream>>>(context, message, emb, Xf);

  gru_kernel<<<(NTOT / 16) * 2, 256, 0, stream>>>(
      Xf, WihF, WhhF, WihB, WhhB, b_ih_f, b_hh_f, b_ih_b, b_hh_b,
      ctx_len, msg_len, hcat);

  outproj_kernel<<<NTOT / 16, 256, 0, stream>>>(hcat, WoutH, b_out, enc);
  att_score_kernel<<<NCTX / 8, 256, 0, stream>>>(enc, w_att, b_att, sc);
  pool_kernel<<<BATCH, 256, 0, stream>>>(enc, sc, seg_len, out);
}
